// AEGRU_12051678233404
// MI455X (gfx1250) — matrix operand order repaired and checked by reading, not compiled, not run on hardware
//
#include <hip/hip_runtime.h>

// ---------------------------------------------------------------------------
// GRU autoencoder for MI455X (gfx1250, wave32).
//   B=128, S=1024, D=128, H=512.
// Strategy:
//   * bf16 operands + v_wmma_f32_16x16x32_bf16 (fp32 accumulate) for all GEMMs
//   * ONE persistent kernel per recurrence (encoder: both layers fused per
//     step; decoder: cell0 / cell1 / projection phases) with a device-wide
//     software barrier between phases -> no per-step launch overhead.
//   * Each block's weight slice (~163-180KB) is staged into the 320KB LDS
//     ONCE, then all B operands come from conflict-free ds_load_b128 for all
//     1024 steps (the slice exceeds the 64KB WGP$, so global reads would miss
//     to L2 every step). A operands (h state) stay global: they are produced
//     by other blocks between barriers.
// ---------------------------------------------------------------------------

#define BB 128
#define SS 1024
#define DD 128
#define HH 512
#define H3 1536
#define PAD 8          // bf16 elements of row padding: row stride = K+8
                       // -> 260 dwords -> +4 banks/row -> minimal conflicts

typedef __bf16 bf16;
typedef __attribute__((ext_vector_type(8)))  __bf16 v8bf;
typedef __attribute__((ext_vector_type(16))) __bf16 v16bf;
typedef __attribute__((ext_vector_type(8)))  float  v8f;

// ------------------------------ WMMA helpers -------------------------------

__device__ __forceinline__ v8f wmma_bf16(v16bf a, v16bf b, v8f c) {
  // D = A(16x32 bf16) * B(32x16 bf16) + C(16x16 f32)
  return __builtin_amdgcn_wmma_f32_16x16x32_bf16(
      /*neg_a=*/false, a, /*neg_b=*/false, b,
      /*c_mod=*/(short)0, c, /*reuse_a=*/false, /*reuse_b=*/false);
}

__device__ __forceinline__ v16bf pack2(v8bf lo, v8bf hi) {
  v16bf r;
#pragma unroll
  for (int i = 0; i < 8; ++i) { r[i] = lo[i]; r[i + 8] = hi[i]; }
  return r;
}

// A-matrix (16x32 bf16) per-lane load, row-major source.
// Lane L (m = L&15, khalf = L>>4) holds K = khalf*8 + {0..7} and
// K = 16 + khalf*8 + {0..7}  (ISA 7.12.2 16-bit A layout).
__device__ __forceinline__ v16bf load_a16(const bf16* rowp, int khalf) {
  v8bf lo = *(const v8bf*)(rowp + khalf * 8);
  v8bf hi = *(const v8bf*)(rowp + 16 + khalf * 8);
  return pack2(lo, hi);
}

// B-matrix (32x16 bf16) per-lane load. B[k][n] = W[j0+n][k]: lane L
// (n = L&15, khalf = L>>4) holds K = khalf*8 + {0..7} and
// K = 16 + khalf*8 + {0..7} of a W row (here a padded LDS row).
__device__ __forceinline__ v16bf load_b16(const bf16* rowp, int khalf) {
  v8bf lo = *(const v8bf*)(rowp + khalf * 8);
  v8bf hi = *(const v8bf*)(rowp + 16 + khalf * 8);
  return pack2(lo, hi);
}

__device__ __forceinline__ float sigmoidf_(float x) {
  return 1.0f / (1.0f + __expf(-x));
}

// --------------------------- device-wide barrier ---------------------------

__device__ __forceinline__ void grid_barrier(unsigned* bar, unsigned* gen,
                                             unsigned nwg) {
  __threadfence();   // make this thread's stores device-visible
  __syncthreads();
  if (threadIdx.x == 0) {
    volatile unsigned* vgen = (volatile unsigned*)gen;
    unsigned g = *vgen;
    if (atomicAdd(bar, 1u) == nwg - 1u) {
      atomicExch(bar, 0u);
      __threadfence();
      atomicAdd(gen, 1u);                 // release
    } else {
      while (*vgen == g) __builtin_amdgcn_s_sleep(1);
    }
  }
  __syncthreads();
  __threadfence();   // acquire: invalidate stale near caches
}

// ------------------------------ weight staging -----------------------------
// Copy nrows rows of W (row-major, row length K) starting at row0 into a
// padded LDS block (row stride K+PAD). One-time cost per persistent kernel.
__device__ void stage_rows(const bf16* __restrict__ W, int K, int row0,
                           int nrows, bf16* __restrict__ dst) {
  const int ld = K + PAD;
  const int vpr = K / 8;                       // v8bf chunks per row
  for (int idx = threadIdx.x; idx < nrows * vpr; idx += blockDim.x) {
    int r = idx / vpr, c = idx - r * vpr;
    *(v8bf*)(dst + (size_t)r * ld + (size_t)c * 8) =
        *(const v8bf*)(W + (size_t)(row0 + r) * K + (size_t)c * 8);
  }
}

// Stage the 3-gate slice {g*H + j0 .. +15} for g=0,1,2 -> 48 padded LDS rows.
__device__ void stage_gates(const bf16* __restrict__ W, int K, int j0,
                            bf16* __restrict__ dst) {
#pragma unroll
  for (int g = 0; g < 3; ++g)
    stage_rows(W, K, g * HH + j0, 16, dst + (size_t)(g * 16) * (K + PAD));
}

// ------------------------------ GRU cell tile ------------------------------
// One wave computes a 16x16 tile of h' = GRUCell(Ain_t, Hprev).
//   gi = Ain @ Wih^T + bih ; gh = Hprev @ Whh^T + bhh
//   r = sig(gir+ghr); z = sig(giz+ghz); n = tanh(gin + r*ghn)
//   h' = (1-z)*n + z*h          (n's gh part kept in its own accumulator)
// B operands come from the per-block LDS weight slice (48 x (K+PAD) rows,
// row g*16+col for gate g).
__device__ void gru_cell_tile(const bf16* __restrict__ Ain, size_t ldaA,
                              int Kin,
                              const bf16* __restrict__ Hprev,
                              const bf16* __restrict__ sWih,  // LDS 48x(Kin+PAD)
                              const bf16* __restrict__ sWhh,  // LDS 48x(HH+PAD)
                              const float* __restrict__ bih,
                              const float* __restrict__ bhh,
                              int b0, int j0, int lane,
                              bf16* __restrict__ Hout) {
  const int m = lane & 15;
  const int khalf = lane >> 4;
  const int col = m;
  const int jr = j0 + col, jz = HH + j0 + col, jn = 2 * HH + j0 + col;

  const float br = bih[jr] + bhh[jr];
  const float bz = bih[jz] + bhh[jz];
  const float bni = bih[jn];
  const float bnh = bhh[jn];

  v8f accr, accz, accni, accnh;
#pragma unroll
  for (int i = 0; i < 8; ++i) {
    accr[i] = br; accz[i] = bz; accni[i] = bni; accnh[i] = bnh;
  }

  // ---- input contribution: gi = Ain @ Wih^T (B from LDS) ----
  const int ldi = Kin + PAD;
  const bf16* arow = Ain + (size_t)(b0 + m) * ldaA;
  const bf16* wr = sWih + (size_t)(0 * 16 + col) * ldi;
  const bf16* wz = sWih + (size_t)(1 * 16 + col) * ldi;
  const bf16* wn = sWih + (size_t)(2 * 16 + col) * ldi;
#pragma unroll 2
  for (int k0 = 0; k0 < Kin; k0 += 32) {
    v16bf a = load_a16(arow + k0, khalf);
    accr  = wmma_bf16(a, load_b16(wr + k0, khalf), accr);
    accz  = wmma_bf16(a, load_b16(wz + k0, khalf), accz);
    accni = wmma_bf16(a, load_b16(wn + k0, khalf), accni);
  }

  // ---- hidden contribution: gh = Hprev @ Whh^T (B from LDS) ----
  const int ldh = HH + PAD;
  const bf16* hrow = Hprev + (size_t)(b0 + m) * HH;
  const bf16* ur = sWhh + (size_t)(0 * 16 + col) * ldh;
  const bf16* uz = sWhh + (size_t)(1 * 16 + col) * ldh;
  const bf16* un = sWhh + (size_t)(2 * 16 + col) * ldh;
#pragma unroll 2
  for (int k0 = 0; k0 < HH; k0 += 32) {
    v16bf a = load_a16(hrow + k0, khalf);
    accr  = wmma_bf16(a, load_b16(ur + k0, khalf), accr);
    accz  = wmma_bf16(a, load_b16(uz + k0, khalf), accz);
    accnh = wmma_bf16(a, load_b16(un + k0, khalf), accnh);
  }

  // ---- gates + state update (C layout: lane half -> rows khalf*8 + r) ----
  const int rb = b0 + khalf * 8;
#pragma unroll
  for (int r = 0; r < 8; ++r) {
    const int row = rb + r;
    float rg = sigmoidf_(accr[r]);
    float zg = sigmoidf_(accz[r]);
    float ng = tanhf(accni[r] + rg * accnh[r]);
    float hold = (float)Hprev[(size_t)row * HH + j0 + col];
    Hout[(size_t)row * HH + j0 + col] = (bf16)((1.0f - zg) * ng + zg * hold);
  }
}

// ------------------------------- small kernels -----------------------------

__global__ void k_f32_to_bf16(const float* __restrict__ s,
                              bf16* __restrict__ d, int n) {
  int i = blockIdx.x * blockDim.x + threadIdx.x;
  int st = gridDim.x * blockDim.x;
  for (; i < n; i += st) d[i] = (bf16)s[i];
}

__global__ void k_init(bf16* h0, bf16* h1, bf16* pred, unsigned* bar) {
  int i = blockIdx.x * blockDim.x + threadIdx.x;
  int st = gridDim.x * blockDim.x;
  for (int k = i; k < BB * HH; k += st) { h0[k] = (bf16)0.0f; h1[k] = (bf16)0.0f; }
  for (int k = i; k < BB * DD; k += st) pred[k] = (bf16)0.0f;
  if (i == 0) { bar[0] = 0u; bar[1] = 0u; }
}

__global__ void k_emit_z(const bf16* __restrict__ h1, float* __restrict__ z) {
  int i = blockIdx.x * blockDim.x + threadIdx.x;
  if (i < BB * HH) z[i] = (float)h1[i];
}

// LDS sizes (bf16 elements)
#define SZ_IH0 (48 * (DD + PAD))
#define SZ_HH  (48 * (HH + PAD))
#define SZ_OUT (16 * (HH + PAD))

// ----------------------------- persistent encoder --------------------------
// 32 blocks x 256 threads (8 waves). Block -> hidden-column tile j0, wave ->
// batch-row tile b0. Both layers fused per timestep; seq1 never hits memory.
__global__ __launch_bounds__(256) void k_encoder(
    const bf16* __restrict__ xbf,                       // [B,S,D] bf16
    const bf16* Wih0, const bf16* Whh0, const float* bih0, const float* bhh0,
    const bf16* Wih1, const bf16* Whh1, const float* bih1, const float* bhh1,
    bf16* h0buf, bf16* h1buf,                           // each 2*B*H (dbl buf)
    unsigned* bar) {
  extern __shared__ __align__(16) char smem_raw[];
  bf16* sWih0 = (bf16*)smem_raw;
  bf16* sWhh0 = sWih0 + SZ_IH0;
  bf16* sWih1 = sWhh0 + SZ_HH;
  bf16* sWhh1 = sWih1 + SZ_HH;

  const int lane = threadIdx.x & 31;
  const int wave = threadIdx.x >> 5;
  const int b0 = wave * 16;
  const int j0 = blockIdx.x * 16;
  const unsigned nwg = gridDim.x;

  // one-time weight staging into LDS (reused for all 1024 steps)
  stage_gates(Wih0, DD, j0, sWih0);
  stage_gates(Whh0, HH, j0, sWhh0);
  stage_gates(Wih1, HH, j0, sWih1);
  stage_gates(Whh1, HH, j0, sWhh1);
  __syncthreads();

  for (int t = 0; t < SS; ++t) {
    const int cur = t & 1, nxt = (t + 1) & 1;
    const bf16* h0r = h0buf + cur * (BB * HH);
    bf16*       h0w = h0buf + nxt * (BB * HH);
    const bf16* h1r = h1buf + cur * (BB * HH);
    bf16*       h1w = h1buf + nxt * (BB * HH);

    if (t + 1 < SS)  // pull next timestep's x rows toward the WGP early
      __builtin_prefetch(
          xbf + (size_t)(b0 + (lane & 15)) * (SS * DD) + (size_t)(t + 1) * DD,
          0, 0);

    // layer 0: h0_t = cell(x_t, h0_{t-1})
    gru_cell_tile(xbf + (size_t)t * DD, (size_t)SS * DD, DD, h0r, sWih0, sWhh0,
                  bih0, bhh0, b0, j0, lane, h0w);
    grid_barrier(bar, bar + 1, nwg);
    // layer 1: h1_t = cell(h0_t, h1_{t-1})
    gru_cell_tile(h0w, HH, HH, h1r, sWih1, sWhh1, bih1, bhh1, b0, j0, lane,
                  h1w);
    grid_barrier(bar, bar + 1, nwg);
  }
}

// ----------------------------- persistent decoder --------------------------
__global__ __launch_bounds__(256) void k_decoder(
    const bf16* Wih0, const bf16* Whh0, const float* bih0, const float* bhh0,
    const bf16* Wih1, const bf16* Whh1, const float* bih1, const float* bhh1,
    const bf16* Wout, const float* bout,                // [D,H], [D]
    bf16* h0buf, bf16* h1buf, bf16* predbuf,            // dbl buffered
    float* __restrict__ out,                            // [B,S,D] fp32
    unsigned* bar) {
  extern __shared__ __align__(16) char smem_raw[];
  bf16* sWih0 = (bf16*)smem_raw;
  bf16* sWhh0 = sWih0 + SZ_IH0;
  bf16* sWih1 = sWhh0 + SZ_HH;
  bf16* sWhh1 = sWih1 + SZ_HH;
  bf16* sWout = sWhh1 + SZ_HH;

  const int lane = threadIdx.x & 31;
  const int wave = threadIdx.x >> 5;
  const int b0 = wave * 16;
  const int j0 = blockIdx.x * 16;
  const int m = lane & 15;
  const int khalf = lane >> 4;
  const int col = m;
  const unsigned nwg = gridDim.x;

  stage_gates(Wih0, DD, j0, sWih0);
  stage_gates(Whh0, HH, j0, sWhh0);
  stage_gates(Wih1, HH, j0, sWih1);
  stage_gates(Whh1, HH, j0, sWhh1);
  if (blockIdx.x < 8)  // projection slice: rows p0..p0+15 of Wout [D,H]
    stage_rows(Wout, HH, blockIdx.x * 16, 16, sWout);
  __syncthreads();

  for (int t = 0; t < SS; ++t) {
    const int cur = t & 1, nxt = (t + 1) & 1;
    const bf16* pr = predbuf + cur * (BB * DD);
    bf16*       pw = predbuf + nxt * (BB * DD);
    const bf16* h0r = h0buf + cur * (BB * HH);
    bf16*       h0w = h0buf + nxt * (BB * HH);
    const bf16* h1r = h1buf + cur * (BB * HH);
    bf16*       h1w = h1buf + nxt * (BB * HH);

    // phase A: h0_t = cell(pred_t, h0_{t-1})
    gru_cell_tile(pr, DD, DD, h0r, sWih0, sWhh0, bih0, bhh0, b0, j0, lane,
                  h0w);
    grid_barrier(bar, bar + 1, nwg);
    // phase B: h1_t = cell(h0_t, h1_{t-1})
    gru_cell_tile(h0w, HH, HH, h1r, sWih1, sWhh1, bih1, bhh1, b0, j0, lane,
                  h1w);
    grid_barrier(bar, bar + 1, nwg);
    // phase C: pred_{t+1} = h1_t @ Wout^T + bout  (64 tiles -> blocks 0..7)
    if (blockIdx.x < 8) {
      const int p0 = blockIdx.x * 16;
      v8f acc;
      const float bb = bout[p0 + col];
#pragma unroll
      for (int i = 0; i < 8; ++i) acc[i] = bb;
      const bf16* hrow = h1w + (size_t)(b0 + m) * HH;
      const bf16* wo = sWout + (size_t)col * (HH + PAD);
#pragma unroll 2
      for (int k0 = 0; k0 < HH; k0 += 32) {
        v16bf a = load_a16(hrow + k0, khalf);
        acc = wmma_bf16(a, load_b16(wo + k0, khalf), acc);
      }
      const int rb = b0 + khalf * 8;
#pragma unroll
      for (int r = 0; r < 8; ++r) {
        const int row = rb + r;
        float v = acc[r];
        out[(size_t)row * SS * DD + (size_t)t * DD + p0 + col] = v;
        pw[(size_t)row * DD + p0 + col] = (bf16)v;
      }
    }
    grid_barrier(bar, bar + 1, nwg);
  }
}

// --------------------------------- launcher --------------------------------

extern "C" void kernel_launch(void* const* d_in, const int* in_sizes, int n_in,
                              void* d_out, int out_size, void* d_ws,
                              size_t ws_size, hipStream_t stream) {
  const float* x      = (const float*)d_in[0];
  const float* eWih0  = (const float*)d_in[1];
  const float* eWhh0  = (const float*)d_in[2];
  const float* ebih0  = (const float*)d_in[3];
  const float* ebhh0  = (const float*)d_in[4];
  const float* eWih1  = (const float*)d_in[5];
  const float* eWhh1  = (const float*)d_in[6];
  const float* ebih1  = (const float*)d_in[7];
  const float* ebhh1  = (const float*)d_in[8];
  const float* dWih0  = (const float*)d_in[9];
  const float* dWhh0  = (const float*)d_in[10];
  const float* dbih0  = (const float*)d_in[11];
  const float* dbhh0  = (const float*)d_in[12];
  const float* dWih1  = (const float*)d_in[13];
  const float* dWhh1  = (const float*)d_in[14];
  const float* dbih1  = (const float*)d_in[15];
  const float* dbhh1  = (const float*)d_in[16];
  const float* fWout  = (const float*)d_in[17];
  const float* fbout  = (const float*)d_in[18];

  float* out = (float*)d_out;                 // [B,S,D] then z [B,H]
  float* zout = out + (size_t)BB * SS * DD;

  // workspace carve-up (256B aligned), total ~45 MB
  char* ws = (char*)d_ws;
  size_t off = 0;
  auto take = [&](size_t bytes) -> char* {
    char* p = ws + off;
    off += (bytes + 255) & ~(size_t)255;
    return p;
  };
  unsigned* bar   = (unsigned*)take(256);
  bf16* xbf       = (bf16*)take((size_t)BB * SS * DD * 2);
  bf16* wEih0     = (bf16*)take((size_t)H3 * DD * 2);
  bf16* wEhh0     = (bf16*)take((size_t)H3 * HH * 2);
  bf16* wEih1     = (bf16*)take((size_t)H3 * HH * 2);
  bf16* wEhh1     = (bf16*)take((size_t)H3 * HH * 2);
  bf16* wDih0     = (bf16*)take((size_t)H3 * DD * 2);
  bf16* wDhh0     = (bf16*)take((size_t)H3 * HH * 2);
  bf16* wDih1     = (bf16*)take((size_t)H3 * HH * 2);
  bf16* wDhh1     = (bf16*)take((size_t)H3 * HH * 2);
  bf16* wOut      = (bf16*)take((size_t)DD * HH * 2);
  bf16* h0buf     = (bf16*)take((size_t)2 * BB * HH * 2);
  bf16* h1buf     = (bf16*)take((size_t)2 * BB * HH * 2);
  bf16* predbuf   = (bf16*)take((size_t)2 * BB * DD * 2);

  auto cvt = [&](const float* s, bf16* d, int n) {
    int blocks = (n + 255) / 256;
    if (blocks > 4096) blocks = 4096;
    k_f32_to_bf16<<<blocks, 256, 0, stream>>>(s, d, n);
  };
  cvt(x, xbf, BB * SS * DD);
  cvt(eWih0, wEih0, H3 * DD);
  cvt(eWhh0, wEhh0, H3 * HH);
  cvt(eWih1, wEih1, H3 * HH);
  cvt(eWhh1, wEhh1, H3 * HH);
  cvt(dWih0, wDih0, H3 * DD);
  cvt(dWhh0, wDhh0, H3 * HH);
  cvt(dWih1, wDih1, H3 * HH);
  cvt(dWhh1, wDhh1, H3 * HH);
  cvt(fWout, wOut, DD * HH);

  k_init<<<256, 256, 0, stream>>>(h0buf, h1buf, predbuf, bar);

  // dynamic LDS: encoder 162816 B, decoder 179456 B (<< 320KB per WGP)
  const size_t encSmem = (size_t)(SZ_IH0 + 3 * SZ_HH) * sizeof(bf16);
  const size_t decSmem = encSmem + (size_t)SZ_OUT * sizeof(bf16);

  // encoder: 32 hidden-column tiles -> 32 persistent workgroups
  k_encoder<<<32, 256, encSmem, stream>>>(xbf, wEih0, wEhh0, ebih0, ebhh0,
                                          wEih1, wEhh1, ebih1, ebhh1, h0buf,
                                          h1buf, bar);

  // z = final encoder layer-1 hidden state (buffer index 0 since S is even)
  k_emit_z<<<(BB * HH + 255) / 256, 256, 0, stream>>>(h1buf, zout);

  // decoder continues from encoder finals in-place (h0buf/h1buf index 0)
  k_decoder<<<32, 256, decSmem, stream>>>(wDih0, wDhh0, dbih0, dbhh0, wDih1,
                                          wDhh1, dbih1, dbhh1, wOut, fbout,
                                          h0buf, h1buf, predbuf, out, bar);
}
